// TernaryWeight_48180943126805
// MI455X (gfx1250) — compile-verified, matrix operand order checked by reading, not run on hardware
//
#include <hip/hip_runtime.h>

typedef __attribute__((ext_vector_type(16))) _Float16 v16h;
typedef __attribute__((ext_vector_type(8)))  _Float16 v8h;
typedef __attribute__((ext_vector_type(8)))  float    v8f;
typedef __attribute__((ext_vector_type(4)))  int      v4i;

// gfx1250 async global->LDS path (ASYNCcnt-tracked), with VGPR-staging fallback.
#if defined(__gfx1250__) && \
    __has_builtin(__builtin_amdgcn_global_load_async_to_lds_b128) && \
    __has_builtin(__builtin_amdgcn_s_wait_asynccnt)
#define USE_ASYNC_LDS 1
#else
#define USE_ASYNC_LDS 0
#endif

__device__ __forceinline__ void copy16_to_lds(const _Float16* gsrc, _Float16* ldst) {
#if USE_ASYNC_LDS
    __builtin_amdgcn_global_load_async_to_lds_b128(
        (__attribute__((address_space(1))) v4i*)(gsrc),
        (__attribute__((address_space(3))) v4i*)(ldst), 0, 0);
#else
    *(v8h*)ldst = *(const v8h*)gsrc;
#endif
}

__device__ __forceinline__ void wait_lds_fill() {
#if USE_ASYNC_LDS
    __builtin_amdgcn_s_wait_asynccnt(0);
#endif
}

// ---------------------------------------------------------------------------
// Kernel 1: per-output-row ternary quantization of W (fp32) -> f16 (q * scale)
// ---------------------------------------------------------------------------
__global__ __launch_bounds__(256) void quantize_weight_kernel(
    const float* __restrict__ W, const float* __restrict__ scale,
    _Float16* __restrict__ Q, int K)
{
    __shared__ float red[256];
    const int o = blockIdx.x;
    const float* w = W + (size_t)o * K;

    float s = 0.0f;
    for (int k = threadIdx.x; k < K; k += 256) s += fabsf(w[k]);
    red[threadIdx.x] = s;
    __syncthreads();
    #pragma unroll
    for (int off = 128; off > 0; off >>= 1) {
        if (threadIdx.x < off) red[threadIdx.x] += red[threadIdx.x + off];
        __syncthreads();
    }
    const float alpha = red[0] / (float)K;
    const float sc = scale[o];

    _Float16* q = Q + (size_t)o * K;
    for (int k = threadIdx.x; k < K; k += 256) {
        float v = w[k];
        float t = (v > alpha) ? 1.0f : ((v < -alpha) ? -1.0f : 0.0f);
        q[k] = (_Float16)(t * sc);
    }
}

// ---------------------------------------------------------------------------
// Kernel 2: bulk fp32 -> f16 convert of activations (n multiple of 8)
// ---------------------------------------------------------------------------
__global__ __launch_bounds__(256) void convert_f32_to_f16_kernel(
    const float* __restrict__ X, _Float16* __restrict__ Y, long n)
{
    long i = ((long)blockIdx.x * 256 + threadIdx.x) * 8;
    if (i + 8 <= n) {
        float4 a = *(const float4*)(X + i);
        float4 b = *(const float4*)(X + i + 4);
        v8h o;
        o[0] = (_Float16)a.x; o[1] = (_Float16)a.y;
        o[2] = (_Float16)a.z; o[3] = (_Float16)a.w;
        o[4] = (_Float16)b.x; o[5] = (_Float16)b.y;
        o[6] = (_Float16)b.z; o[7] = (_Float16)b.w;
        *(v8h*)(Y + i) = o;
    }
}

// ---------------------------------------------------------------------------
// Kernel 3: GEMM  out[M,N] = A[M,K](f16) * Bq[N,K]^T(f16) + bias (f32 accum)
//   Block tile 128(M) x 256(N) x 32(K), 256 threads = 8 waves (2M x 4N).
//   Wave tile 64x64 = 4x4 v_wmma_f32_16x16x32_f16 accumulators.
//   Double-buffered LDS (48 KB), async global->LDS fill when available.
// ---------------------------------------------------------------------------
__device__ __forceinline__ v16h frag_load(const _Float16* p) {
    // ISA 16-bit fragment layout: lanes 0-15 K=0..7 & 16..23, lanes 16-31
    // K=8..15 & 24..31 -> two 16B LDS reads per fragment.
    v8h lo = *(const v8h*)p;
    v8h hi = *(const v8h*)(p + 16);
    v16h r;
    #pragma unroll
    for (int i = 0; i < 8; ++i) { r[i] = lo[i]; r[i + 8] = hi[i]; }
    return r;
}

__global__ __launch_bounds__(256) void ternary_gemm_wmma_kernel(
    const _Float16* __restrict__ A,   // M x K, row-major (f16 activations)
    const _Float16* __restrict__ Bq,  // N x K, row-major (ternary*scale f16)
    const float* __restrict__ bias,
    float* __restrict__ out,
    int M, int N, int K)
{
    __shared__ _Float16 sA[2][128 * 32];   // 2 x 8 KB
    __shared__ _Float16 sB[2][256 * 32];   // 2 x 16 KB

    const int tid  = threadIdx.x;
    const int lane = tid & 31;
    const int wave = tid >> 5;

    const int m0 = blockIdx.y * 128;
    const int n0 = blockIdx.x * 256;
    const int wm = (wave >> 2) * 64;     // 0 / 64
    const int wn = (wave & 3) * 64;      // 0 / 64 / 128 / 192

    const int fr = lane & 15;            // row within 16x16 fragment
    const int fk = (lane >> 4) << 3;     // 0 or 8 (half index)

    // cooperative-load chunk coordinates (16B chunks of a row-major [rows][32] tile)
    const int crow = tid >> 2;           // 0..63
    const int ccol = (tid & 3) << 3;     // 0,8,16,24

    v8f acc[4][4];
    #pragma unroll
    for (int mi = 0; mi < 4; ++mi)
        #pragma unroll
        for (int ni = 0; ni < 4; ++ni)
            acc[mi][ni] = (v8f){0.f,0.f,0.f,0.f,0.f,0.f,0.f,0.f};

    // ---- tile loader: A 128x32 (512 chunks), B 256x32 (1024 chunks) ----
    auto load_tiles = [&](int k0, int buf) {
        #pragma unroll
        for (int r = 0; r < 2; ++r) {
            const int row = crow + r * 64;
            copy16_to_lds(&A[(size_t)(m0 + row) * K + k0 + ccol],
                          &sA[buf][row * 32 + ccol]);
        }
        #pragma unroll
        for (int r = 0; r < 4; ++r) {
            const int row = crow + r * 64;
            copy16_to_lds(&Bq[(size_t)(n0 + row) * K + k0 + ccol],
                          &sB[buf][row * 32 + ccol]);
        }
    };

    load_tiles(0, 0);
    wait_lds_fill();
    __syncthreads();

    int cur = 0;
    for (int k0 = 0; k0 < K; k0 += 32) {
        if (k0 + 32 < K) load_tiles(k0 + 32, cur ^ 1);

        const _Float16* bufA = &sA[cur][0];
        const _Float16* bufB = &sB[cur][0];

        v16h af[4], bf[4];
        #pragma unroll
        for (int mi = 0; mi < 4; ++mi)
            af[mi] = frag_load(&bufA[(wm + mi * 16 + fr) * 32 + fk]);
        #pragma unroll
        for (int ni = 0; ni < 4; ++ni)
            bf[ni] = frag_load(&bufB[(wn + ni * 16 + fr) * 32 + fk]);

        #pragma unroll
        for (int mi = 0; mi < 4; ++mi)
            #pragma unroll
            for (int ni = 0; ni < 4; ++ni)
                acc[mi][ni] = __builtin_amdgcn_wmma_f32_16x16x32_f16(
                    /*neg_a=*/false, af[mi], /*neg_b=*/false, bf[ni],
                    /*c_mod=*/(short)0, acc[mi][ni],
                    /*reuse_a=*/false, /*reuse_b=*/false);

        wait_lds_fill();     // next buffer fully written (async path)
        __syncthreads();     // all waves done reading cur / writing cur^1
        cur ^= 1;
    }

    // epilogue: D layout = 8 VGPRs; lanes 0-15 -> M=v, N=lane;
    //           lanes 16-31 -> M=v+8, N=lane-16
    const int mhi = (lane >> 4) << 3;
    #pragma unroll
    for (int mi = 0; mi < 4; ++mi) {
        #pragma unroll
        for (int ni = 0; ni < 4; ++ni) {
            const int n = n0 + wn + ni * 16 + fr;
            const float bv = bias[n];
            #pragma unroll
            for (int v = 0; v < 8; ++v) {
                const int m = m0 + wm + mi * 16 + mhi + v;
                out[(size_t)m * N + n] = acc[mi][ni][v] + bv;
            }
        }
    }
}

// ---------------------------------------------------------------------------
// Launch
// ---------------------------------------------------------------------------
extern "C" void kernel_launch(void* const* d_in, const int* in_sizes, int n_in,
                              void* d_out, int out_size, void* d_ws, size_t ws_size,
                              hipStream_t stream) {
    const float* x      = (const float*)d_in[0];   // (B,S,DIN) f32
    const float* weight = (const float*)d_in[1];   // (DOUT,DIN) f32
    const float* scale  = (const float*)d_in[2];   // (DOUT,1) f32
    const float* bias   = (const float*)d_in[3];   // (DOUT,) f32
    float* out = (float*)d_out;

    const int  DOUT = in_sizes[3];                  // 4096
    const int  DIN  = in_sizes[1] / DOUT;           // 4096
    const long MX   = (long)in_sizes[0] / DIN;      // B*S = 8192

    _Float16* x_f16 = (_Float16*)d_ws;
    _Float16* qw    = (_Float16*)((char*)d_ws + (size_t)MX * DIN * sizeof(_Float16));

    // 1) quantize weights -> f16 ternary*scale, [N][K]
    quantize_weight_kernel<<<DOUT, 256, 0, stream>>>(weight, scale, qw, DIN);

    // 2) convert activations -> f16
    const long nx = MX * (long)DIN;
    convert_f32_to_f16_kernel<<<(int)(nx / (256 * 8)), 256, 0, stream>>>(x, x_f16, nx);

    // 3) WMMA GEMM with bias epilogue
    dim3 grid(DOUT / 256, (int)(MX / 128));
    ternary_gemm_wmma_kernel<<<grid, 256, 0, stream>>>(x_f16, qw, bias, out,
                                                       (int)MX, DOUT, DIN);
}
